// LSHAttention_63754494542110
// MI455X (gfx1250) — matrix operand order repaired and checked by reading, not compiled, not run on hardware
//
#include <hip/hip_runtime.h>
#include <hip/hip_bf16.h>

#define DEVI __device__ __forceinline__

typedef __attribute__((ext_vector_type(2)))  float  v2f;
typedef __attribute__((ext_vector_type(8)))  float  v8f;
typedef __attribute__((ext_vector_type(16))) __bf16 v16bf;

constexpr int B_ = 4, L_ = 8192, D_ = 512, H_ = 8, HD_ = 64;
constexpr int NBUCK = 128, NR_ = 4, CS_ = 64, NC_ = L_ / CS_;
constexpr int BL_ = B_ * L_;
constexpr float NEGV = -1e9f;

// ---------------- WMMA wrappers ----------------
DEVI v8f wmma_bf16(v16bf a, v16bf b, v8f c) {
  return __builtin_amdgcn_wmma_f32_16x16x32_bf16(false, a, false, b, (short)0, c, false, false);
}
DEVI v8f wmma_f32(v2f a, v2f b, v8f c) {
  return __builtin_amdgcn_wmma_f32_16x16x4_f32(false, a, false, b, (short)0, c, false, false);
}

// ---------------- bf16 fragment loaders (ISA 7.12.2 layouts) ----------------
// A: 16(M) x 32(K), row-major src [M][ld]. lane&15 = M; halves split K.
DEVI v16bf frag_A(const __bf16* src, int ld, int m0, int k0, int lane) {
  int hl = (lane >> 4) & 1, m = lane & 15;
  const __bf16* row = src + (size_t)(m0 + m) * ld;
  v16bf a;
#pragma unroll
  for (int j = 0; j < 8; ++j) {
    int kb = k0 + 2 * j + 8 * hl + ((j >= 4) ? 8 : 0);
    a[2 * j]     = row[kb];
    a[2 * j + 1] = row[kb + 1];
  }
  return a;
}
// B: 32(K) x 16(N), row-major src [K][ld]. lane&15 = N; halves split K in 8-blocks (0-7,16-23 / 8-15,24-31).
DEVI v16bf frag_B(const __bf16* src, int ld, int k0, int n0, int lane) {
  int hl = (lane >> 4) & 1, n = lane & 15;
  v16bf b;
#pragma unroll
  for (int j = 0; j < 8; ++j) {
    int kb = k0 + 2 * j + 8 * hl + ((j >= 4) ? 8 : 0);
    b[2 * j]     = src[(size_t)kb * ld + n0 + n];
    b[2 * j + 1] = src[(size_t)(kb + 1) * ld + n0 + n];
  }
  return b;
}
// B fragment from transposed storage [N][ld] (K pairs contiguous in a row).
DEVI v16bf frag_BT(const __bf16* src, int ld, int k0, int n0, int lane) {
  int hl = (lane >> 4) & 1, n = lane & 15;
  const __bf16* row = src + (size_t)(n0 + n) * ld;
  v16bf b;
#pragma unroll
  for (int j = 0; j < 8; ++j) {
    int kb = k0 + 2 * j + 8 * hl + ((j >= 4) ? 8 : 0);
    b[2 * j]     = row[kb];
    b[2 * j + 1] = row[kb + 1];
  }
  return b;
}

// ---------------- f32 -> bf16 convert ----------------
__global__ void cvt_kernel(const float* __restrict__ s, __bf16* __restrict__ d, int n) {
  int i = blockIdx.x * 256 + threadIdx.x;
  if (i < n) d[i] = (__bf16)s[i];
}

// ---------------- QK projection: fp32 WMMA + bias + per-head L2 norm ----------------
// Wave computes a 16-row x 64-col (one head) strip. Output layout qkf[b][h][l][hd].
__global__ __launch_bounds__(128) void proj_qk_kernel(const float* __restrict__ x,
                                                      const float* __restrict__ Wqk,
                                                      const float* __restrict__ bqk,
                                                      float* __restrict__ qkf) {
  int lane = threadIdx.x & 31, w = threadIdx.x >> 5;
  int gw = blockIdx.x * 4 + w;
  int strip = gw & 7;          // head
  int rt = gw >> 3;            // row tile of BL
  int row0 = rt * 16, col0 = strip * 64;
  int hl = (lane >> 4) & 1, ln = lane & 15;
  v8f C[4] = {};
  for (int k0 = 0; k0 < D_; k0 += 4) {
    v2f a;
    const float* xr = x + (size_t)(row0 + ln) * D_ + k0 + 2 * hl;
    a.x = xr[0]; a.y = xr[1];
#pragma unroll
    for (int nt = 0; nt < 4; ++nt) {
      v2f bb;
      const float* wp = Wqk + (size_t)(k0 + 2 * hl) * D_ + col0 + nt * 16 + ln;
      bb.x = wp[0]; bb.y = wp[D_];
      C[nt] = wmma_f32(a, bb, C[nt]);
    }
  }
  float bias[4];
#pragma unroll
  for (int nt = 0; nt < 4; ++nt) bias[nt] = bqk[col0 + nt * 16 + ln];
#pragma unroll
  for (int r_ = 0; r_ < 8; ++r_) {
    float cv[4], s = 0.f;
#pragma unroll
    for (int nt = 0; nt < 4; ++nt) { cv[nt] = C[nt][r_] + bias[nt]; s += cv[nt] * cv[nt]; }
#pragma unroll
    for (int off = 1; off < 16; off <<= 1) s += __shfl_xor(s, off, 32);
    float invn = 1.f / fmaxf(sqrtf(s), 1e-12f);
    int row = row0 + r_ + 8 * hl;
    int b_ = row >> 13, l = row & (L_ - 1);
    float* dst = qkf + (((size_t)(b_ * H_ + strip)) * L_ + l) * HD_;
#pragma unroll
    for (int nt = 0; nt < 4; ++nt) dst[nt * 16 + ln] = cv[nt] * invn;
  }
}

// ---------------- V projection: bf16 WMMA ----------------
__global__ __launch_bounds__(128) void proj_v_kernel(const __bf16* __restrict__ xh,
                                                     const __bf16* __restrict__ Wvh,
                                                     const float* __restrict__ bv,
                                                     float* __restrict__ vf) {
  int lane = threadIdx.x & 31, w = threadIdx.x >> 5;
  int gw = blockIdx.x * 4 + w;
  int strip = gw & 7, rt = gw >> 3;
  int row0 = rt * 16, col0 = strip * 64;
  int hl = (lane >> 4) & 1, ln = lane & 15;
  v8f C[4] = {};
  for (int k0 = 0; k0 < D_; k0 += 32) {
    v16bf a = frag_A(xh, D_, row0, k0, lane);
#pragma unroll
    for (int nt = 0; nt < 4; ++nt) {
      v16bf bb = frag_B(Wvh, D_, k0, col0 + nt * 16, lane);
      C[nt] = wmma_bf16(a, bb, C[nt]);
    }
  }
  float bias[4];
#pragma unroll
  for (int nt = 0; nt < 4; ++nt) bias[nt] = bv[col0 + nt * 16 + ln];
#pragma unroll
  for (int r_ = 0; r_ < 8; ++r_) {
    int row = row0 + r_ + 8 * hl;
    int b_ = row >> 13, l = row & (L_ - 1);
    float* dst = vf + (((size_t)(b_ * H_ + strip)) * L_ + l) * HD_;
#pragma unroll
    for (int nt = 0; nt < 4; ++nt) dst[nt * 16 + ln] = C[nt][r_] + bias[nt];
  }
}

// ---------------- Hash: fp32 WMMA (qk @ rot) + first-occurrence argmax of [R;-R] ----------------
__global__ __launch_bounds__(128) void hash_kernel(const float* __restrict__ qkf,
                                                   const float* __restrict__ rot,
                                                   int* __restrict__ buckets) {
  __shared__ float R[4][16][64];
  int lane = threadIdx.x & 31, w = threadIdx.x >> 5;
  int gw = blockIdx.x * 4 + w;
  int tg = gw & (L_ / 16 - 1);
  int tmp = gw >> 9;
  int h = tmp & 7; tmp >>= 3;
  int b_ = tmp & 3; int r = tmp >> 2;
  int hl = (lane >> 4) & 1, ln = lane & 15;
  const float* Q  = qkf + (((size_t)(b_ * H_ + h)) * L_ + tg * 16) * HD_;
  const float* Rt = rot + ((size_t)(r * H_ + h)) * HD_ * 64;
  v8f C[4] = {};
  for (int k0 = 0; k0 < HD_; k0 += 4) {
    v2f a;
    const float* qr = Q + (size_t)ln * HD_ + k0 + 2 * hl;
    a.x = qr[0]; a.y = qr[1];
#pragma unroll
    for (int nt = 0; nt < 4; ++nt) {
      v2f bb;
      const float* wp = Rt + (size_t)(k0 + 2 * hl) * 64 + nt * 16 + ln;
      bb.x = wp[0]; bb.y = wp[64];
      C[nt] = wmma_f32(a, bb, C[nt]);
    }
  }
#pragma unroll
  for (int nt = 0; nt < 4; ++nt)
#pragma unroll
    for (int r_ = 0; r_ < 8; ++r_)
      R[w][r_ + 8 * hl][nt * 16 + ln] = C[nt][r_];
  __syncthreads();
  if (lane < 16) {
    const float* row = R[w][lane];
    float best = -3.4e38f; int bj = 0;
    for (int j = 0; j < 64; ++j)  { float v =  row[j]; if (v > best) { best = v; bj = j; } }
    for (int j = 0; j < 64; ++j)  { float v = -row[j]; if (v > best) { best = v; bj = 64 + j; } }
    buckets[((size_t)((r * B_ + b_) * H_ + h)) * L_ + tg * 16 + lane] = bj;
  }
}

// ---------------- Stable counting sort by bucket, per (round,b,h) ----------------
__global__ __launch_bounds__(128) void sort_kernel(const int* __restrict__ buckets,
                                                   int* __restrict__ sidx,
                                                   int* __restrict__ bsort) {
  __shared__ unsigned short hist[NBUCK * 128];   // [bucket][thread]
  __shared__ int base[NBUCK];
  int t = threadIdx.x;
  size_t gbase = (size_t)blockIdx.x * L_;
  const int* bk = buckets + gbase;
  for (int bb = 0; bb < NBUCK; ++bb) hist[bb * 128 + t] = 0;
  __syncthreads();
  int seg0 = t * 64;
  for (int i = 0; i < 64; ++i) hist[bk[seg0 + i] * 128 + t]++;
  __syncthreads();
  { // per-bucket exclusive prefix over thread segments (thread t owns bucket t)
    int run = 0;
    for (int tt = 0; tt < 128; ++tt) { int c = hist[t * 128 + tt]; hist[t * 128 + tt] = (unsigned short)run; run += c; }
    base[t] = run;
  }
  __syncthreads();
  if (t == 0) { int run = 0; for (int bb = 0; bb < NBUCK; ++bb) { int c = base[bb]; base[bb] = run; run += c; } }
  __syncthreads();
  int* so = sidx + gbase; int* sb = bsort + gbase;
  for (int i = 0; i < 64; ++i) {
    int idx = seg0 + i, bb = bk[idx];
    int pos = base[bb] + hist[bb * 128 + t];
    hist[bb * 128 + t]++;
    so[pos] = idx; sb[pos] = bb;
  }
}

// ---------------- Chunked LSH attention for one round, scatter-add into acc ----------------
__global__ __launch_bounds__(128) void attn_kernel(const float* __restrict__ qkf,
                                                   const float* __restrict__ vf,
                                                   const int* __restrict__ sidx,
                                                   const int* __restrict__ bsort,
                                                   float* __restrict__ acc,
                                                   int r, int addFlag) {
  __shared__ __bf16 Qs[64 * 64];     // sorted queries
  __shared__ __bf16 Ks[128 * 64];    // keys (prev chunk + cur chunk)
  __shared__ __bf16 Vt[64 * 128];    // values transposed [hd][key]
  __shared__ __bf16 Ps[64 * 128];    // unnormalized probs
  __shared__ float  invl[64];
  __shared__ int    bq_s[64], bk_s[128], oidx[64];
  int tid = threadIdx.x;
  int lane = tid & 31, w = tid >> 5;
  int hl = (lane >> 4) & 1, ln = lane & 15;
  int c = blockIdx.x;
  int bh = blockIdx.y;
  int b_ = bh >> 3, h = bh & 7;
  size_t sbase = ((size_t)((r * B_ + b_) * H_ + h)) * L_;
  size_t hbase = ((size_t)(b_ * H_ + h)) * L_;
  { // gather key row j = tid (j<64: previous chunk; j>=64: current chunk)
    int j = tid, kpos = c * 64 - 64 + j;
    if (kpos < 0) {
      bk_s[j] = -1;
      for (int d = 0; d < 64; ++d) Ks[j * 64 + d] = (__bf16)0.f;
      for (int d = 0; d < 64; ++d) Vt[d * 128 + j] = (__bf16)0.f;
    } else {
      int ok = sidx[sbase + kpos];
      bk_s[j] = bsort[sbase + kpos];
      const float* qr = qkf + (hbase + ok) * HD_;
      const float* vr = vf  + (hbase + ok) * HD_;
      for (int d = 0; d < 64; ++d) Ks[j * 64 + d] = (__bf16)qr[d];
      for (int d = 0; d < 64; ++d) Vt[d * 128 + j] = (__bf16)vr[d];
    }
  }
  if (tid < 64) { // gather query row
    int i = tid, spos = c * 64 + i;
    int oi = sidx[sbase + spos];
    oidx[i] = oi;
    bq_s[i] = bsort[sbase + spos];
    const float* qr = qkf + (hbase + oi) * HD_;
    for (int d = 0; d < 64; ++d) Qs[i * 64 + d] = (__bf16)qr[d];
  }
  __syncthreads();
  // S = Q @ K^T : wave w covers query rows w*16..w*16+15, all 128 keys
  v8f S[8] = {};
  for (int k0 = 0; k0 < 64; k0 += 32) {
    v16bf a = frag_A(Qs, 64, w * 16, k0, lane);
#pragma unroll
    for (int nt = 0; nt < 8; ++nt) {
      v16bf bb = frag_BT(Ks, 64, k0, nt * 16, lane);
      S[nt] = wmma_bf16(a, bb, S[nt]);
    }
  }
  // masked softmax (unnormalized), matching reference semantics
#pragma unroll
  for (int r_ = 0; r_ < 8; ++r_) {
    int q = w * 16 + r_ + 8 * hl;
    int bqv = bq_s[q];
    float pv[8]; float m = NEGV;
#pragma unroll
    for (int nt = 0; nt < 8; ++nt) {
      int j = nt * 16 + ln;
      int bkv = bk_s[j];
      bool msk = (bkv != bqv) | (bkv < 0) | (j > q + 64);
      float s = msk ? NEGV : S[nt][r_] * 0.125f;   // / sqrt(64)
      pv[nt] = s; m = fmaxf(m, s);
    }
#pragma unroll
    for (int off = 1; off < 16; off <<= 1) m = fmaxf(m, __shfl_xor(m, off, 32));
    float sum = 0.f;
#pragma unroll
    for (int nt = 0; nt < 8; ++nt) {
      int j = nt * 16 + ln;
      int bkv = bk_s[j];
      bool msk = (bkv != bqv) | (bkv < 0) | (j > q + 64);
      float p = msk ? 0.f : __expf(pv[nt] - m);
      pv[nt] = p; sum += p;
    }
#pragma unroll
    for (int off = 1; off < 16; off <<= 1) sum += __shfl_xor(sum, off, 32);
    float inv = sum > 0.f ? 1.f / sum : 0.f;       // all-masked row -> exact zeros
    if (ln == 0) invl[q] = inv;
#pragma unroll
    for (int nt = 0; nt < 8; ++nt) Ps[q * 128 + nt * 16 + ln] = (__bf16)pv[nt];
  }
  __syncthreads();
  // O = P @ V
  v8f O[4] = {};
  for (int k0 = 0; k0 < 128; k0 += 32) {
    v16bf a = frag_A(Ps, 128, w * 16, k0, lane);
#pragma unroll
    for (int nt = 0; nt < 4; ++nt) {
      v16bf bb = frag_BT(Vt, 128, k0, nt * 16, lane);
      O[nt] = wmma_bf16(a, bb, O[nt]);
    }
  }
  // normalize rows and scatter-add back to original positions (permutation: race-free)
#pragma unroll
  for (int r_ = 0; r_ < 8; ++r_) {
    int q = w * 16 + r_ + 8 * hl;
    float inv = invl[q];
    float* dst = acc + (hbase + oidx[q]) * HD_;
#pragma unroll
    for (int nt = 0; nt < 4; ++nt) {
      int hd = nt * 16 + ln;
      float val = O[nt][r_] * inv;
      if (addFlag) val += dst[hd];
      dst[hd] = val;
    }
  }
}

// ---------------- Output projection: mean over rounds folded into bf16 A-fragment ----------------
__global__ __launch_bounds__(128) void out_proj_kernel(const float* __restrict__ acc,
                                                       const __bf16* __restrict__ Woh,
                                                       const float* __restrict__ bo_,
                                                       float* __restrict__ out) {
  int lane = threadIdx.x & 31, w = threadIdx.x >> 5;
  int gw = blockIdx.x * 4 + w;
  int strip = gw & 7, rt = gw >> 3;
  int row0 = rt * 16, col0 = strip * 64;
  int hl = (lane >> 4) & 1, ln = lane & 15;
  int row = row0 + ln;                 // A-row for this lane
  int b_ = row >> 13, l = row & (L_ - 1);
  v8f C[4] = {};
  for (int k0 = 0; k0 < D_; k0 += 32) {
    v16bf a;
#pragma unroll
    for (int j = 0; j < 8; ++j) {
      int kb = k0 + 2 * j + 8 * hl + ((j >= 4) ? 8 : 0);
      int hh = kb >> 6, hd = kb & 63;  // kb even => kb,kb+1 in same head
      const float* ar = acc + (((size_t)(b_ * H_ + hh)) * L_ + l) * HD_ + hd;
      a[2 * j]     = (__bf16)(0.25f * ar[0]);
      a[2 * j + 1] = (__bf16)(0.25f * ar[1]);
    }
#pragma unroll
    for (int nt = 0; nt < 4; ++nt) {
      v16bf bb = frag_B(Woh, D_, k0, col0 + nt * 16, lane);
      C[nt] = wmma_bf16(a, bb, C[nt]);
    }
  }
  float bias[4];
#pragma unroll
  for (int nt = 0; nt < 4; ++nt) bias[nt] = bo_[col0 + nt * 16 + ln];
#pragma unroll
  for (int r_ = 0; r_ < 8; ++r_) {
    int orow = row0 + r_ + 8 * hl;
#pragma unroll
    for (int nt = 0; nt < 4; ++nt)
      out[(size_t)orow * D_ + col0 + nt * 16 + ln] = C[nt][r_] + bias[nt];
  }
}

// ---------------- Host launch ----------------
extern "C" void kernel_launch(void* const* d_in, const int* in_sizes, int n_in,
                              void* d_out, int out_size, void* d_ws, size_t ws_size,
                              hipStream_t stream) {
  const float* x   = (const float*)d_in[0];
  const float* Wqk = (const float*)d_in[1];
  const float* bqk = (const float*)d_in[2];
  const float* Wv  = (const float*)d_in[3];
  const float* bv  = (const float*)d_in[4];
  const float* Wo  = (const float*)d_in[5];
  const float* bo  = (const float*)d_in[6];
  const float* rot = (const float*)d_in[7];
  float* out = (float*)d_out;

  char* wptr = (char*)d_ws;
  auto carve = [&](size_t bytes) { char* p = wptr; wptr += (bytes + 255) & ~(size_t)255; return p; };
  float*  qkf     = (float*)carve((size_t)B_ * H_ * L_ * HD_ * 4);
  float*  vf      = (float*)carve((size_t)B_ * H_ * L_ * HD_ * 4);
  float*  acc     = (float*)carve((size_t)B_ * H_ * L_ * HD_ * 4);
  __bf16* xh      = (__bf16*)carve((size_t)BL_ * D_ * 2);
  __bf16* Wvh     = (__bf16*)carve((size_t)D_ * D_ * 2);
  __bf16* Woh     = (__bf16*)carve((size_t)D_ * D_ * 2);
  int*    buckets = (int*)carve((size_t)NR_ * B_ * H_ * L_ * 4);
  int*    sidx    = (int*)carve((size_t)NR_ * B_ * H_ * L_ * 4);
  int*    bsort   = (int*)carve((size_t)NR_ * B_ * H_ * L_ * 4);
  (void)ws_size; (void)in_sizes; (void)n_in; (void)out_size;

  cvt_kernel<<<(BL_ * D_ + 255) / 256, 256, 0, stream>>>(x, xh, BL_ * D_);
  cvt_kernel<<<(D_ * D_ + 255) / 256, 256, 0, stream>>>(Wv, Wvh, D_ * D_);
  cvt_kernel<<<(D_ * D_ + 255) / 256, 256, 0, stream>>>(Wo, Woh, D_ * D_);

  proj_qk_kernel<<<4096, 128, 0, stream>>>(x, Wqk, bqk, qkf);
  proj_v_kernel<<<4096, 128, 0, stream>>>(xh, Wvh, bv, vf);
  hash_kernel<<<16384, 128, 0, stream>>>(qkf, rot, buckets);
  sort_kernel<<<NR_ * B_ * H_, 128, 0, stream>>>(buckets, sidx, bsort);
  for (int r = 0; r < NR_; ++r)
    attn_kernel<<<dim3(NC_, B_ * H_), 128, 0, stream>>>(qkf, vf, sidx, bsort, acc, r, r > 0);
  out_proj_kernel<<<4096, 128, 0, stream>>>(acc, Woh, bo, out);
}
